// Encoder_31379031064599
// MI455X (gfx1250) — matrix operand order repaired and checked by reading, not compiled, not run on hardware
//
#include <hip/hip_runtime.h>
#include <hip/hip_bf16.h>

// ---- problem constants ----
#define HDIM   256
#define K2     512          // concat K = H(x) + H(h)
#define G4     1024         // 4*H gates
#define BATCH  1024
#define TSTEPS 64
#define IMGD   2048
#define VOCABN 32000
#define BT     32           // batch rows per workgroup (32 WGs total)

// ---- workspace layout (bytes), total ~23.7 MB ----
#define OFF_EMB    0u                         // VOCAB*H bf16     = 16,384,000
#define OFF_WCAT   16384000u                  // G4*K2 bf16       =  1,048,576
#define OFF_WIMG   17432576u                  // H*IMGD bf16      =  1,048,576
#define OFF_IMGBF  18481152u                  // B*IMGD bf16      =  4,194,304
#define OFF_IMGOUT 22675456u                  // B*H f32          =  1,048,576

#if __has_builtin(__builtin_amdgcn_global_load_async_to_lds_b128) && \
    __has_builtin(__builtin_amdgcn_s_wait_asynccnt)
#define HAVE_ASYNC_LDS 1
#else
#define HAVE_ASYNC_LDS 0
#endif

typedef __attribute__((ext_vector_type(16))) __bf16 v16bf;
typedef __attribute__((ext_vector_type(8)))  float  v8f;
typedef __attribute__((ext_vector_type(4)))  int    v4i;

union bfpack { uint4 u[2]; v16bf v; };

// A-matrix 16x32 bf16 tile (ISA 7.12.2): lane m=L%16; first 8 elems K=(L/16)*8+0..7,
// last 8 elems K=16+(L/16)*8+0..7. Source row-major [M][K], row stride ldk elems.
__device__ __forceinline__ v16bf load_tileA(const __bf16* p0, int ldk, int lane) {
  const __bf16* p = p0 + (lane & 15) * ldk + ((lane >> 4) << 3);
  bfpack u;
  u.u[0] = *(const uint4*)(p);
  u.u[1] = *(const uint4*)(p + 16);
  return u.v;
}
// B-matrix 32x16 bf16 tile: lane n=L%16; first 8 elems K=(L/16)*8+0..7, last 8 elems K=16+(L/16)*8+0..7.
// gates = X * W^T with W row-major [N][K], so B[k][n] = W[n][k].
__device__ __forceinline__ v16bf load_tileB(const __bf16* p0, int ldk, int lane) {
  const __bf16* p = p0 + (lane & 15) * ldk + ((lane >> 4) << 3);
  bfpack u;
  u.u[0] = *(const uint4*)(p);
  u.u[1] = *(const uint4*)(p + 16);
  return u.v;
}

__device__ __forceinline__ v8f wmma_bf16(v16bf a, v16bf b, v8f c) {
  return __builtin_amdgcn_wmma_f32_16x16x32_bf16(false, a, false, b, (short)0, c,
                                                 false, false);
}

__device__ __forceinline__ float sigmoidf(float x) {
  return 1.0f / (1.0f + __expf(-x));
}

// async (double-buffered) embedding gather: 8 threads/row, 4x B128 each
__device__ __forceinline__ void gather_x(const __bf16* __restrict__ emb,
                                         const int* __restrict__ cap,
                                         int t, int bbase, int tid,
                                         __bf16* __restrict__ xdst) {
  int r   = tid >> 3;                  // 0..31
  int col = (tid & 7) * 32;
  int token = cap[t * BATCH + bbase + r];
  const __bf16* src = emb + (size_t)token * HDIM + col;
  __bf16* dst = xdst + r * HDIM + col;
#if HAVE_ASYNC_LDS
#pragma unroll
  for (int j = 0; j < 4; ++j) {
    __builtin_amdgcn_global_load_async_to_lds_b128(
        (v4i*)(const_cast<__bf16*>(src) + j * 8),
        (v4i*)(dst + j * 8),
        0, 0);
  }
#else
  const uint4* s = (const uint4*)src;
  uint4*       d = (uint4*)dst;
  d[0] = s[0]; d[1] = s[1]; d[2] = s[2]; d[3] = s[3];
#endif
}

__device__ __forceinline__ void wait_async_gather() {
#if HAVE_ASYNC_LDS
  __builtin_amdgcn_s_wait_asynccnt(0);
#endif
}

// ---------------- prep: fp32 -> bf16 ----------------
__global__ void k_f32_to_bf16(const float* __restrict__ src,
                              __bf16* __restrict__ dst, int n) {
  int i = blockIdx.x * blockDim.x + threadIdx.x;
  if (i < n) dst[i] = (__bf16)src[i];
}

// Wcat[g][k] = k<H ? W_ih[g][k] : W_hh[g][k-H]   (bf16, row-major [1024][512])
__global__ void k_build_wcat(const float* __restrict__ Wih,
                             const float* __restrict__ Whh,
                             __bf16* __restrict__ dst) {
  int i = blockIdx.x * blockDim.x + threadIdx.x;   // over 1024*512
  int g = i >> 9;
  int k = i & 511;
  float v = (k < HDIM) ? Wih[g * HDIM + k] : Whh[g * HDIM + (k - HDIM)];
  dst[i] = (__bf16)v;
}

// ---------------- image GEMM: imgout = relu(imgbf * Wimg^T + b) ----------------
__global__ void k_img_gemm(const __bf16* __restrict__ A,     // [1024][2048]
                           const __bf16* __restrict__ W,     // [256][2048]
                           const float* __restrict__ bias,   // [256]
                           float* __restrict__ out) {        // [1024][256]
  int tid  = threadIdx.x;
  int lane = tid & 31;
  int gid  = blockIdx.x * 8 + (tid >> 5);   // 0..1023 -> one 16x16 tile per wave
  int Mb   = (gid >> 4) * 16;
  int Nb   = (gid & 15) * 16;
  v8f acc = {0.f, 0.f, 0.f, 0.f, 0.f, 0.f, 0.f, 0.f};
#pragma unroll 4
  for (int kk = 0; kk < IMGD / 32; ++kk) {
    v16bf a = load_tileA(A + (size_t)Mb * IMGD + kk * 32, IMGD, lane);
    v16bf b = load_tileB(W + (size_t)Nb * IMGD + kk * 32, IMGD, lane);
    acc = wmma_bf16(a, b, acc);
  }
  int col = Nb + (lane & 15);
  float bi = bias[col];
#pragma unroll
  for (int v = 0; v < 8; ++v) {
    int row = Mb + v + 8 * (lane >> 4);
    out[row * HDIM + col] = fmaxf(acc[v] + bi, 0.0f);
  }
}

// ---------------- fused LSTM over T=64 steps ----------------
// grid = 32 WGs (32 batch rows each), block = 256 threads (8 waves).
// LDS: xbuf[2][32][256] bf16 | hbuf[32][256] bf16 | c f32[32][256]
//      gates f32[32][1024] | imgadd f32[32][256]  -> 240 KB total.
__global__ void k_lstm(const __bf16* __restrict__ emb,   // [VOCAB][256] bf16
                       const int* __restrict__ cap,      // [64][1024]
                       const __bf16* __restrict__ Wcat,  // [1024][512] bf16
                       const float* __restrict__ b_ih,   // [1024]
                       const float* __restrict__ b_hh,   // [1024]
                       const float* __restrict__ imgout, // [1024][256]
                       float* __restrict__ out) {        // [64][1024][256]
  extern __shared__ __align__(16) char smem[];
  __bf16* xbuf0  = (__bf16*)(smem);                 // 16384 B
  __bf16* xbuf1  = (__bf16*)(smem + 16384);         // 16384 B
  __bf16* hbuf   = (__bf16*)(smem + 32768);         // 16384 B
  float*  cst    = (float*)(smem + 49152);          // 32768 B
  float*  gates  = (float*)(smem + 81920);          // 131072 B
  float*  imgadd = (float*)(smem + 212992);         // 32768 B

  const int tid   = threadIdx.x;                    // hc = tid (0..255)
  const int lane  = tid & 31;
  const int wv    = tid >> 5;                       // 0..7
  const int bbase = blockIdx.x * BT;
  const int Nwave = wv * 128;                       // this wave's gate columns

  // hoisted per-thread biases (gate order i,f,g,o; column hc = tid)
  const float bi_i = b_ih[tid]            + b_hh[tid];
  const float bi_f = b_ih[HDIM + tid]     + b_hh[HDIM + tid];
  const float bi_g = b_ih[2 * HDIM + tid] + b_hh[2 * HDIM + tid];
  const float bi_o = b_ih[3 * HDIM + tid] + b_hh[3 * HDIM + tid];

  // prologue: async-gather x(0); init h, c, imgadd
  gather_x(emb, cap, 0, bbase, tid, xbuf0);
#pragma unroll
  for (int i = 0; i < 32; ++i) {
    hbuf[i * HDIM + tid] = (__bf16)0.0f;
    cst[i * HDIM + tid]  = 0.0f;
    imgadd[i * HDIM + tid] = imgout[(bbase + i) * HDIM + tid];
  }
  wait_async_gather();
  __syncthreads();

  for (int t = 0; t < TSTEPS; ++t) {
    const __bf16* xcur = (t & 1) ? xbuf1 : xbuf0;
    // prefetch next step's embeddings into the other buffer (async -> LDS)
    if (t + 1 < TSTEPS)
      gather_x(emb, cap, t + 1, bbase, tid, (t & 1) ? xbuf0 : xbuf1);

    // ---- GEMM: gates[32x1024] = [x|h][32x512] * Wcat^T ----
    {
      v8f acc0[8], acc1[8];
      const v8f z = {0.f, 0.f, 0.f, 0.f, 0.f, 0.f, 0.f, 0.f};
#pragma unroll
      for (int n = 0; n < 8; ++n) { acc0[n] = z; acc1[n] = z; }
      // x half: K = 0..255
#pragma unroll 2
      for (int kk = 0; kk < 8; ++kk) {
        v16bf a0 = load_tileA(xcur + kk * 32, HDIM, lane);
        v16bf a1 = load_tileA(xcur + 16 * HDIM + kk * 32, HDIM, lane);
#pragma unroll
        for (int nt = 0; nt < 8; ++nt) {
          v16bf b = load_tileB(Wcat + (size_t)(Nwave + nt * 16) * K2 + kk * 32,
                               K2, lane);
          acc0[nt] = wmma_bf16(a0, b, acc0[nt]);
          acc1[nt] = wmma_bf16(a1, b, acc1[nt]);
        }
      }
      // h half: K = 256..511
#pragma unroll 2
      for (int kk = 0; kk < 8; ++kk) {
        v16bf a0 = load_tileA(hbuf + kk * 32, HDIM, lane);
        v16bf a1 = load_tileA(hbuf + 16 * HDIM + kk * 32, HDIM, lane);
#pragma unroll
        for (int nt = 0; nt < 8; ++nt) {
          v16bf b = load_tileB(Wcat + (size_t)(Nwave + nt * 16) * K2 + HDIM +
                               kk * 32, K2, lane);
          acc0[nt] = wmma_bf16(a0, b, acc0[nt]);
          acc1[nt] = wmma_bf16(a1, b, acc1[nt]);
        }
      }
      int cl = lane & 15, rh = 8 * (lane >> 4);
#pragma unroll
      for (int nt = 0; nt < 8; ++nt) {
        int col = Nwave + nt * 16 + cl;
#pragma unroll
        for (int v = 0; v < 8; ++v) {
          gates[(v + rh) * G4 + col]      = acc0[nt][v];
          gates[(16 + v + rh) * G4 + col] = acc1[nt][v];
        }
      }
    }
    __syncthreads();

    // ---- cell update + output (h + img); r = i, hc = tid ----
#pragma unroll 4
    for (int i = 0; i < 32; ++i) {
      float gi = gates[i * G4 + tid]            + bi_i;
      float gf = gates[i * G4 + HDIM + tid]     + bi_f;
      float gg = gates[i * G4 + 2 * HDIM + tid] + bi_g;
      float go = gates[i * G4 + 3 * HDIM + tid] + bi_o;
      float fi = sigmoidf(gi);
      float ff = sigmoidf(gf);
      float fg = tanhf(gg);
      float fo = sigmoidf(go);
      float cn = ff * cst[i * HDIM + tid] + fi * fg;
      cst[i * HDIM + tid] = cn;
      float hn = fo * tanhf(cn);
      hbuf[i * HDIM + tid] = (__bf16)hn;
      out[((size_t)t * BATCH + bbase + i) * HDIM + tid] = hn + imgadd[i * HDIM + tid];
    }
    wait_async_gather();     // x(t+1) fully in LDS before next GEMM
    __syncthreads();
  }
}

extern "C" void kernel_launch(void* const* d_in, const int* in_sizes, int n_in,
                              void* d_out, int out_size, void* d_ws, size_t ws_size,
                              hipStream_t stream) {
  const float* img_f = (const float*)d_in[0];   // [1024][2048]
  const int*   cap   = (const int*)d_in[1];     // [64][1024]
  const float* W_img = (const float*)d_in[2];   // [256][2048]
  const float* b_img = (const float*)d_in[3];   // [256]
  const float* emb_f = (const float*)d_in[4];   // [32000][256]
  const float* W_ih  = (const float*)d_in[5];   // [1024][256]
  const float* W_hh  = (const float*)d_in[6];   // [1024][256]
  const float* b_ih  = (const float*)d_in[7];   // [1024]
  const float* b_hh  = (const float*)d_in[8];   // [1024]

  char* ws = (char*)d_ws;                       // needs ~23.7 MB
  __bf16* emb_bf  = (__bf16*)(ws + OFF_EMB);
  __bf16* wcat    = (__bf16*)(ws + OFF_WCAT);
  __bf16* wimg_bf = (__bf16*)(ws + OFF_WIMG);
  __bf16* img_bf  = (__bf16*)(ws + OFF_IMGBF);
  float*  imgout  = (float*)(ws + OFF_IMGOUT);

  k_f32_to_bf16<<<(VOCABN * HDIM + 255) / 256, 256, 0, stream>>>(emb_f, emb_bf,
                                                                 VOCABN * HDIM);
  k_f32_to_bf16<<<(HDIM * IMGD + 255) / 256, 256, 0, stream>>>(W_img, wimg_bf,
                                                               HDIM * IMGD);
  k_f32_to_bf16<<<(BATCH * IMGD + 255) / 256, 256, 0, stream>>>(img_f, img_bf,
                                                                BATCH * IMGD);
  k_build_wcat<<<(G4 * K2) / 256, 256, 0, stream>>>(W_ih, W_hh, wcat);

  k_img_gemm<<<128, 256, 0, stream>>>(img_bf, wimg_bf, b_img, imgout);

  // fused LSTM + broadcast add (WMMA + async LDS gather), 240KB dynamic LDS/WG
  k_lstm<<<BATCH / BT, 256, 245760, stream>>>(emb_bf, cap, wcat, b_ih, b_hh,
                                              imgout, (float*)d_out);
}
